// gwnet_81295140979294
// MI455X (gfx1250) — matrix operand order repaired and checked by reading, not compiled, not run on hardware
//
#include <hip/hip_runtime.h>
#include <cstdint>
#include <cstddef>

// ---------------- model constants ----------------
#define NNODE 1024
#define BATCH 4
#define CH    32      // residual / dilation channels
#define SCCH  256     // skip channels
#define TMAX  64
#define NLAYER 24

// ---------------- vector types ----------------
typedef float          v8f   __attribute__((ext_vector_type(8)));
typedef __bf16         v16bf __attribute__((ext_vector_type(16)));
typedef unsigned short u16x8 __attribute__((ext_vector_type(8)));
typedef unsigned short u16x4 __attribute__((ext_vector_type(4)));

struct U16Pair { u16x8 lo, hi; };

static __device__ __forceinline__ v16bf pack_bf(u16x8 lo, u16x8 hi) {
  U16Pair p{lo, hi};
  return __builtin_bit_cast(v16bf, p);
}
static __device__ __forceinline__ unsigned short f2bf(float f) {
  unsigned u = __builtin_bit_cast(unsigned, f);
  unsigned r = u + 0x7FFFu + ((u >> 16) & 1u);   // round-nearest-even
  return (unsigned short)(r >> 16);
}
static __device__ __forceinline__ float bf2f(unsigned short h) {
  unsigned u = ((unsigned)h) << 16;
  return __builtin_bit_cast(float, u);
}

// =====================================================================
// WMMA GEMM:  Out[M,1024] = X[M,1024] @ Adj[1024,1024]   (bf16 in, bf16 out,
// fp32 accumulate via v_wmma_f32_16x16x32_bf16).  M = gridDim.x*128.
// Block = 8 wave32 = 256 thr, tile 128(M) x 64(N), K stepped by 32 via LDS.
// =====================================================================
#define GTM 128
#define GTN 64
#define GKB 32
#define LDA_P 40   // padded LDS row (elems) for X tile rows
#define LDB_P 40   // padded LDS row (elems) for transposed Adj cols

__global__ __launch_bounds__(256) void k_gemm_bf16(
    const unsigned short* __restrict__ Xb,   // [M,1024] bf16 row-major
    const unsigned short* __restrict__ Ab,   // [1024,1024] bf16 row-major
    unsigned short* __restrict__ Ob)         // [M,1024] bf16
{
  __shared__ __align__(16) unsigned short ldsA[GTM * LDA_P];
  __shared__ __align__(16) unsigned short ldsB[GTN * LDB_P];
  const int tid  = threadIdx.x;
  const int lane = tid & 31;
  const int wave = tid >> 5;
  const int wm = wave & 3;          // 4 waves along M
  const int wn = wave >> 2;         // 2 waves along N
  const int m0 = blockIdx.x * GTM;
  const int n0 = blockIdx.y * GTN;
  const int l16   = lane & 15;
  const int halfA = (lane >> 4) * 8;    // A-frag K base: 0 / 8 (+16 for upper 8 elems)
  const int halfB = (lane >> 4) * 8;    // B-frag K base: 0 / 8 (+16 for upper 8 elems)

  v8f acc[2][2];
  const v8f vzero = {0.f,0.f,0.f,0.f,0.f,0.f,0.f,0.f};
  #pragma unroll
  for (int i = 0; i < 2; ++i)
    #pragma unroll
    for (int j = 0; j < 2; ++j) acc[i][j] = vzero;

  for (int kb = 0; kb < 1024; kb += GKB) {
    // stage X tile (128 x 32), row-major
    #pragma unroll
    for (int i = 0; i < 4; ++i) {
      int flat = tid + 256 * i;          // 0..1023 -> 16B chunks
      int row = flat >> 2;
      int kc  = (flat & 3) * 8;
      u16x8 v = *(const u16x8*)(Xb + (size_t)(m0 + row) * 1024 + kb + kc);
      *(u16x8*)(ldsA + row * LDA_P + kc) = v;
    }
    // stage Adj tile (32 x 64) transposed -> ldsB[n][k]
    #pragma unroll
    for (int i = 0; i < 2; ++i) {
      int flat = tid + 256 * i;          // 0..511 quads
      int k  = flat >> 4;
      int nq = (flat & 15) * 4;
      u16x4 v = *(const u16x4*)(Ab + (size_t)(kb + k) * 1024 + n0 + nq);
      #pragma unroll
      for (int j = 0; j < 4; ++j) ldsB[(nq + j) * LDB_P + k] = v[j];
    }
    if (kb + GKB < 1024) {  // gfx1250 global_prefetch_b8 of next K tiles
      __builtin_prefetch(Xb + (size_t)(m0 + (tid >> 1)) * 1024 + kb + GKB, 0, 0);
      __builtin_prefetch(Ab + (size_t)(kb + GKB + lane) * 1024 + n0, 0, 0);
    }
    __syncthreads();

    v16bf afrg[2], bfrg[2];
    #pragma unroll
    for (int mt = 0; mt < 2; ++mt) {
      int row = wm * 32 + mt * 16 + l16;
      u16x8 lo = *(const u16x8*)(ldsA + row * LDA_P + halfA);
      u16x8 hi = *(const u16x8*)(ldsA + row * LDA_P + halfA + 16);
      afrg[mt] = pack_bf(lo, hi);
    }
    #pragma unroll
    for (int nt = 0; nt < 2; ++nt) {
      int col = wn * 32 + nt * 16 + l16;
      u16x8 lo = *(const u16x8*)(ldsB + col * LDB_P + halfB);
      u16x8 hi = *(const u16x8*)(ldsB + col * LDB_P + halfB + 16);
      bfrg[nt] = pack_bf(lo, hi);
    }
    #pragma unroll
    for (int mt = 0; mt < 2; ++mt)
      #pragma unroll
      for (int nt = 0; nt < 2; ++nt)
        acc[mt][nt] = __builtin_amdgcn_wmma_f32_16x16x32_bf16(
            false, afrg[mt], false, bfrg[nt], (short)0, acc[mt][nt], false, false);
    __syncthreads();
  }

  const int rbase = (lane < 16) ? 0 : 8;   // C/D layout: VGPR r -> M = r (+8 upper half)
  #pragma unroll
  for (int mt = 0; mt < 2; ++mt) {
    #pragma unroll
    for (int nt = 0; nt < 2; ++nt) {
      int col  = n0 + wn * 32 + nt * 16 + l16;
      int row0 = m0 + wm * 32 + mt * 16 + rbase;
      #pragma unroll
      for (int r = 0; r < 8; ++r)
        Ob[(size_t)(row0 + r) * 1024 + col] = f2bf(acc[mt][nt][r]);
    }
  }
}

// =====================================================================
// small helper kernels
// =====================================================================
__global__ void k_zero(float* __restrict__ p, size_t n) {
  size_t i = (size_t)blockIdx.x * blockDim.x + threadIdx.x;
  if (i < n) p[i] = 0.f;
}

__global__ void k_f32_to_bf16(const float* __restrict__ src,
                              unsigned short* __restrict__ dst, size_t n) {
  size_t i = (size_t)blockIdx.x * blockDim.x + threadIdx.x;
  if (i < n) dst[i] = f2bf(src[i]);
}

// E[b,n,0:32] = [tod_emb | dow_emb | spatial_emb]
__global__ void k_embed(const float* __restrict__ X,
                        const float* __restrict__ tod_emb,
                        const float* __restrict__ dow_emb,
                        const float* __restrict__ spat_emb,
                        float* __restrict__ E) {
  int idx = blockIdx.x * blockDim.x + threadIdx.x;
  if (idx >= BATCH * NNODE) return;
  int b = idx >> 10, n = idx & 1023;
  int ti = (int)X[((size_t)(b * 3 + 1) * NNODE + n) * TMAX + (TMAX - 1)];
  int di = (int)X[((size_t)(b * 3 + 2) * NNODE + n) * TMAX + (TMAX - 1)];
  ti = ti < 0 ? 0 : (ti > 287 ? 287 : ti);
  di = di < 0 ? 0 : (di > 6 ? 6 : di);
  float* e = E + (size_t)idx * 32;
  #pragma unroll
  for (int j = 0; j < 8; ++j)  e[j]      = tod_emb[ti * 8 + j];
  #pragma unroll
  for (int j = 0; j < 8; ++j)  e[8 + j]  = dow_emb[di * 8 + j];
  #pragma unroll
  for (int j = 0; j < 16; ++j) e[16 + j] = spat_emb[n * 16 + j];
}

// Z[n,m] = relu(nv1[n,:10] . nv2[:10,m])
__global__ void k_adp_logits(const float* __restrict__ nv1,
                             const float* __restrict__ nv2,
                             float* __restrict__ Z) {
  int idx = blockIdx.x * blockDim.x + threadIdx.x;
  if (idx >= NNODE * NNODE) return;
  int n = idx >> 10, m = idx & 1023;
  float s = 0.f;
  #pragma unroll
  for (int k = 0; k < 10; ++k) s += nv1[n * 10 + k] * nv2[k * NNODE + m];
  Z[idx] = fmaxf(s, 0.f);
}

// row-wise softmax, write bf16
__global__ __launch_bounds__(256) void k_row_softmax_bf16(
    const float* __restrict__ Z, unsigned short* __restrict__ Abf) {
  __shared__ float red[256];
  int n = blockIdx.x, tid = threadIdx.x;
  const float* row = Z + (size_t)n * NNODE;
  float mx = -1e30f;
  for (int m = tid; m < NNODE; m += 256) mx = fmaxf(mx, row[m]);
  red[tid] = mx; __syncthreads();
  for (int s = 128; s > 0; s >>= 1) { if (tid < s) red[tid] = fmaxf(red[tid], red[tid + s]); __syncthreads(); }
  mx = red[0]; __syncthreads();
  float sum = 0.f;
  for (int m = tid; m < NNODE; m += 256) sum += expf(row[m] - mx);
  red[tid] = sum; __syncthreads();
  for (int s = 128; s > 0; s >>= 1) { if (tid < s) red[tid] += red[tid + s]; __syncthreads(); }
  float inv = 1.f / red[0];
  for (int m = tid; m < NNODE; m += 256)
    Abf[(size_t)n * NNODE + m] = f2bf(expf(row[m] - mx) * inv);
}

// out[b,c,n] = bias[c] + W[c,:32] . E[b,n,:32]
__global__ void k_film(const float* __restrict__ W, const float* __restrict__ bias,
                       const float* __restrict__ E, float* __restrict__ out, int C) {
  int idx = blockIdx.x * blockDim.x + threadIdx.x;
  if (idx >= BATCH * C * NNODE) return;
  int n = idx & 1023;
  int c = (idx >> 10) % C;
  int b = idx / (C * NNODE);
  const float* e = E + ((size_t)(b * NNODE + n)) * 32;
  float a = bias[c];
  #pragma unroll
  for (int j = 0; j < 32; ++j) a += W[c * 32 + j] * e[j];
  out[idx] = a;
}

// start conv [3->32] + FiLM; h layout [B,32,64,N]
__global__ void k_start(const float* __restrict__ X,
                        const float* __restrict__ sW, const float* __restrict__ sb,
                        const float* __restrict__ rs, const float* __restrict__ rh,
                        float* __restrict__ h) {
  int idx = blockIdx.x * blockDim.x + threadIdx.x;
  if (idx >= BATCH * TMAX * NNODE) return;
  int b = idx / (TMAX * NNODE);
  int r = idx % (TMAX * NNODE);
  int t = r >> 10, n = r & 1023;
  float x0 = X[((size_t)(b * 3 + 0) * NNODE + n) * TMAX + t];
  float x1 = X[((size_t)(b * 3 + 1) * NNODE + n) * TMAX + t];
  float x2 = X[((size_t)(b * 3 + 2) * NNODE + n) * TMAX + t];
  #pragma unroll
  for (int c = 0; c < CH; ++c) {
    float v = sb[c] + sW[c * 3] * x0 + sW[c * 3 + 1] * x1 + sW[c * 3 + 2] * x2;
    size_t fi = ((size_t)(b * CH + c)) * NNODE + n;
    v = v * (1.f + rs[fi]) + rh[fi];
    h[(((size_t)(b * CH + c)) * TMAX + t) * NNODE + n] = v;
  }
}

// gated dilated conv: writes fp32 h_g and bf16 copy (both compact at T_out)
__global__ __launch_bounds__(256) void k_gatedconv(
    const float* __restrict__ hin,
    const float* __restrict__ fW, const float* __restrict__ fb,
    const float* __restrict__ gW, const float* __restrict__ gb,
    float* __restrict__ hg, unsigned short* __restrict__ hgbf,
    int T_in, int T_out, int d) {
  __shared__ float fWl[CH * CH * 2];
  __shared__ float gWl[CH * CH * 2];
  for (int e = threadIdx.x; e < CH * CH * 2; e += 256) { fWl[e] = fW[e]; gWl[e] = gW[e]; }
  __syncthreads();
  int idx = blockIdx.x * blockDim.x + threadIdx.x;
  if (idx >= BATCH * T_out * NNODE) return;
  int b = idx / (T_out * NNODE);
  int r = idx % (T_out * NNODE);
  int t = r >> 10, n = r & 1023;
  float x0[CH], x1[CH];
  #pragma unroll
  for (int ci = 0; ci < CH; ++ci) {
    x0[ci] = hin[(((size_t)(b * CH + ci)) * T_in + t) * NNODE + n];
    x1[ci] = hin[(((size_t)(b * CH + ci)) * T_in + t + d) * NNODE + n];
  }
  for (int c = 0; c < CH; ++c) {
    float f = fb[c], g = gb[c];
    #pragma unroll
    for (int ci = 0; ci < CH; ++ci) {
      f += fWl[(c * CH + ci) * 2] * x0[ci] + fWl[(c * CH + ci) * 2 + 1] * x1[ci];
      g += gWl[(c * CH + ci) * 2] * x0[ci] + gWl[(c * CH + ci) * 2 + 1] * x1[ci];
    }
    float v = tanhf(f) * (1.f / (1.f + expf(-g)));
    size_t o = (((size_t)(b * CH + c)) * T_out + t) * NNODE + n;
    hg[o] = v;
    hgbf[o] = f2bf(v);
  }
}

// skip tail accumulation: skip_acc[b,sc,tt,n] += skb[sc] + skW[sc,:].hg[:, T_out-8+tt]
__global__ __launch_bounds__(256) void k_skipacc(
    const float* __restrict__ hg, const float* __restrict__ skW,
    const float* __restrict__ skb, float* __restrict__ skip_acc, int T_out) {
  __shared__ float Wl[SCCH * CH];
  __shared__ float bl[SCCH];
  for (int e = threadIdx.x; e < SCCH * CH; e += 256) Wl[e] = skW[e];
  for (int e = threadIdx.x; e < SCCH; e += 256) bl[e] = skb[e];
  __syncthreads();
  int idx = blockIdx.x * blockDim.x + threadIdx.x;
  if (idx >= BATCH * 8 * NNODE) return;
  int b = idx / (8 * NNODE);
  int r = idx % (8 * NNODE);
  int tt = r >> 10, n = r & 1023;
  int t = T_out - 8 + tt;
  float hv[CH];
  #pragma unroll
  for (int ci = 0; ci < CH; ++ci)
    hv[ci] = hg[(((size_t)(b * CH + ci)) * T_out + t) * NNODE + n];
  for (int sc = 0; sc < SCCH; ++sc) {
    float s = bl[sc];
    #pragma unroll
    for (int ci = 0; ci < CH; ++ci) s += Wl[sc * CH + ci] * hv[ci];
    skip_acc[(((size_t)(b * SCCH + sc)) * 8 + tt) * NNODE + n] += s;
  }
}

// per-layer 2-element softmax of gcn alphas -> wbuf[4]
__global__ void k_alphas(const float* __restrict__ alpha, float* __restrict__ wbuf) {
  for (int s = 0; s < 2; ++s) {
    float a0 = alpha[s * 2], a1 = alpha[s * 2 + 1];
    float m = fmaxf(a0, a1);
    float e0 = expf(a0 - m), e1 = expf(a1 - m);
    float inv = 1.f / (e0 + e1);
    wbuf[s * 2] = e0 * inv; wbuf[s * 2 + 1] = e1 * inv;
  }
}

// GCN mix pass 1: h_next = bias + beta*W0.hg + w00*W1.xpa + w01*W2.xpb + residual tail
__global__ __launch_bounds__(256) void k_mix1(
    const float* __restrict__ hg,
    const unsigned short* __restrict__ xpa, const unsigned short* __restrict__ xpb,
    const float* __restrict__ W, const float* __restrict__ gbias,
    const float* __restrict__ wbuf, const float* __restrict__ beta_p,
    const float* __restrict__ hres, float* __restrict__ hnext,
    int T_in, int T_out, int d) {
  __shared__ float Wl[CH * 160];
  for (int e = threadIdx.x; e < CH * 160; e += 256) Wl[e] = W[e];
  __syncthreads();
  int idx = blockIdx.x * blockDim.x + threadIdx.x;
  if (idx >= BATCH * T_out * NNODE) return;
  int b = idx / (T_out * NNODE);
  int r = idx % (T_out * NNODE);
  int t = r >> 10, n = r & 1023;
  float beta = beta_p[0], w0 = wbuf[0], w1 = wbuf[1];
  float acc[CH], v[CH];
  #pragma unroll
  for (int co = 0; co < CH; ++co) acc[co] = gbias[co];
  #pragma unroll
  for (int ci = 0; ci < CH; ++ci)
    v[ci] = hg[(((size_t)(b * CH + ci)) * T_out + t) * NNODE + n];
  #pragma unroll
  for (int co = 0; co < CH; ++co) {
    float s = 0.f;
    #pragma unroll
    for (int ci = 0; ci < CH; ++ci) s += Wl[co * 160 + ci] * v[ci];
    acc[co] += beta * s;
  }
  #pragma unroll
  for (int ci = 0; ci < CH; ++ci)
    v[ci] = bf2f(xpa[(((size_t)(b * CH + ci)) * T_out + t) * NNODE + n]);
  #pragma unroll
  for (int co = 0; co < CH; ++co) {
    float s = 0.f;
    #pragma unroll
    for (int ci = 0; ci < CH; ++ci) s += Wl[co * 160 + 32 + ci] * v[ci];
    acc[co] += w0 * s;
  }
  #pragma unroll
  for (int ci = 0; ci < CH; ++ci)
    v[ci] = bf2f(xpb[(((size_t)(b * CH + ci)) * T_out + t) * NNODE + n]);
  #pragma unroll
  for (int co = 0; co < CH; ++co) {
    float s = 0.f;
    #pragma unroll
    for (int ci = 0; ci < CH; ++ci) s += Wl[co * 160 + 64 + ci] * v[ci];
    acc[co] += w1 * s;
  }
  #pragma unroll
  for (int co = 0; co < CH; ++co) {
    acc[co] += hres[(((size_t)(b * CH + co)) * T_in + t + d) * NNODE + n];
    hnext[(((size_t)(b * CH + co)) * T_out + t) * NNODE + n] = acc[co];
  }
}

// GCN mix pass 2: h_next += w10*W3.xpa + w11*W4.xpb
__global__ __launch_bounds__(256) void k_mix2(
    const unsigned short* __restrict__ xpa, const unsigned short* __restrict__ xpb,
    const float* __restrict__ W, const float* __restrict__ wbuf,
    float* __restrict__ hnext, int T_out) {
  __shared__ float Wl[CH * 160];
  for (int e = threadIdx.x; e < CH * 160; e += 256) Wl[e] = W[e];
  __syncthreads();
  int idx = blockIdx.x * blockDim.x + threadIdx.x;
  if (idx >= BATCH * T_out * NNODE) return;
  int b = idx / (T_out * NNODE);
  int r = idx % (T_out * NNODE);
  int t = r >> 10, n = r & 1023;
  float w2 = wbuf[2], w3 = wbuf[3];
  float acc[CH], v[CH];
  #pragma unroll
  for (int co = 0; co < CH; ++co) acc[co] = 0.f;
  #pragma unroll
  for (int ci = 0; ci < CH; ++ci)
    v[ci] = bf2f(xpa[(((size_t)(b * CH + ci)) * T_out + t) * NNODE + n]);
  #pragma unroll
  for (int co = 0; co < CH; ++co) {
    float s = 0.f;
    #pragma unroll
    for (int ci = 0; ci < CH; ++ci) s += Wl[co * 160 + 96 + ci] * v[ci];
    acc[co] += w2 * s;
  }
  #pragma unroll
  for (int ci = 0; ci < CH; ++ci)
    v[ci] = bf2f(xpb[(((size_t)(b * CH + ci)) * T_out + t) * NNODE + n]);
  #pragma unroll
  for (int co = 0; co < CH; ++co) {
    float s = 0.f;
    #pragma unroll
    for (int ci = 0; ci < CH; ++ci) s += Wl[co * 160 + 128 + ci] * v[ci];
    acc[co] += w3 * s;
  }
  #pragma unroll
  for (int co = 0; co < CH; ++co) {
    size_t o = (((size_t)(b * CH + co)) * T_out + t) * NNODE + n;
    hnext[o] += acc[co];
  }
}

// group-norm stats: one block per (b, group); channels per group contiguous
__global__ __launch_bounds__(256) void k_gnstats(
    const float* __restrict__ h, float* __restrict__ stats, int T_out) {
  __shared__ float rs[256], rq[256];
  int b = blockIdx.x >> 3, g = blockIdx.x & 7;
  size_t base = ((size_t)(b * CH + g * 4)) * T_out * NNODE;
  size_t cnt = (size_t)4 * T_out * NNODE;
  float s = 0.f, q = 0.f;
  for (size_t e = threadIdx.x; e < cnt; e += 256) {
    float v = h[base + e];
    s += v; q += v * v;
  }
  rs[threadIdx.x] = s; rq[threadIdx.x] = q; __syncthreads();
  for (int st = 128; st > 0; st >>= 1) {
    if (threadIdx.x < st) { rs[threadIdx.x] += rs[threadIdx.x + st]; rq[threadIdx.x] += rq[threadIdx.x + st]; }
    __syncthreads();
  }
  if (threadIdx.x == 0) {
    float mean = rs[0] / (float)cnt;
    float var = rq[0] / (float)cnt - mean * mean;
    stats[blockIdx.x * 2] = mean;
    stats[blockIdx.x * 2 + 1] = rsqrtf(var + 1e-5f);
  }
}

__global__ void k_gnapply(float* __restrict__ h, const float* __restrict__ stats,
                          const float* __restrict__ gamma, const float* __restrict__ beta,
                          int T_out) {
  size_t idx = (size_t)blockIdx.x * blockDim.x + threadIdx.x;
  size_t total = (size_t)BATCH * CH * T_out * NNODE;
  if (idx >= total) return;
  int c = (int)((idx / ((size_t)T_out * NNODE)) % CH);
  int b = (int)(idx / ((size_t)CH * T_out * NNODE));
  int g = c >> 2;
  float mean = stats[(b * 8 + g) * 2];
  float rstd = stats[(b * 8 + g) * 2 + 1];
  h[idx] = (h[idx] - mean) * rstd * gamma[c] + beta[c];
}

// relu(skip)+FiLM -> st conv [256->32]
__global__ __launch_bounds__(256) void k_stconv(
    const float* __restrict__ skip_acc,
    const float* __restrict__ ss, const float* __restrict__ sh,
    const float* __restrict__ stW, const float* __restrict__ stb,
    float* __restrict__ st) {
  __shared__ float Wl[CH * SCCH];
  for (int e = threadIdx.x; e < CH * SCCH; e += 256) Wl[e] = stW[e];
  __syncthreads();
  int idx = blockIdx.x * blockDim.x + threadIdx.x;
  if (idx >= BATCH * 8 * NNODE) return;
  int b = idx / (8 * NNODE);
  int r = idx % (8 * NNODE);
  int tt = r >> 10, n = r & 1023;
  float acc[CH];
  #pragma unroll
  for (int c2 = 0; c2 < CH; ++c2) acc[c2] = stb[c2];
  for (int sc = 0; sc < SCCH; ++sc) {
    float raw = skip_acc[(((size_t)(b * SCCH + sc)) * 8 + tt) * NNODE + n];
    size_t fi = ((size_t)(b * SCCH + sc)) * NNODE + n;
    float m = fmaxf(raw, 0.f) * (1.f + ss[fi]) + sh[fi];
    #pragma unroll
    for (int c2 = 0; c2 < CH; ++c2) acc[c2] += Wl[c2 * SCCH + sc] * m;
  }
  #pragma unroll
  for (int c2 = 0; c2 < CH; ++c2)
    st[(((size_t)(b * CH + c2)) * 8 + tt) * NNODE + n] = acc[c2];
}

// per-(b,n) cross-attention, fused with output proj, residual and T-mean -> cond
__global__ __launch_bounds__(256) void k_attn(
    const float* __restrict__ hq, const float* __restrict__ st,
    const float* __restrict__ qW, const float* __restrict__ qb,
    const float* __restrict__ kW, const float* __restrict__ kb2,
    const float* __restrict__ vW, const float* __restrict__ vb,
    const float* __restrict__ aoW, const float* __restrict__ aob,
    float* __restrict__ cond) {
  __shared__ float qs[CH * 8], ks[CH * 8];
  __shared__ float Qm[512 * 8], Km[512 * 8], Vm[512 * 8];
  __shared__ float attn[8 * 8 * 8];
  __shared__ float osum[512];
  int b = blockIdx.x >> 10, n = blockIdx.x & 1023;
  int tid = threadIdx.x;
  {
    int c = tid >> 3, t = tid & 7;
    qs[tid] = hq[(((size_t)(b * CH + c)) * 8 + t) * NNODE + n];
    ks[tid] = st[(((size_t)(b * CH + c)) * 8 + t) * NNODE + n];
  }
  __syncthreads();
  for (int e = tid; e < 4096; e += 256) {
    int hid = e >> 3, t = e & 7;
    float aq = qb[hid], ak = kb2[hid], av = vb[hid];
    #pragma unroll
    for (int c = 0; c < CH; ++c) {
      aq += qW[hid * CH + c] * qs[c * 8 + t];
      ak += kW[hid * CH + c] * ks[c * 8 + t];
      av += vW[hid * CH + c] * ks[c * 8 + t];
    }
    Qm[e] = aq; Km[e] = ak; Vm[e] = av;
  }
  __syncthreads();
  if (tid < 64) {
    int h8 = tid >> 3, t = tid & 7;
    float sc[8]; float mx = -1e30f;
    #pragma unroll
    for (int s = 0; s < 8; ++s) {
      float d = 0.f;
      for (int dd = 0; dd < 64; ++dd)
        d += Qm[(h8 * 64 + dd) * 8 + t] * Km[(h8 * 64 + dd) * 8 + s];
      sc[s] = d * 0.125f;          // 1/sqrt(64)
      mx = fmaxf(mx, sc[s]);
    }
    float sum = 0.f;
    #pragma unroll
    for (int s = 0; s < 8; ++s) { sc[s] = expf(sc[s] - mx); sum += sc[s]; }
    float inv = 1.f / sum;
    #pragma unroll
    for (int s = 0; s < 8; ++s) attn[(h8 * 8 + t) * 8 + s] = sc[s] * inv;
  }
  __syncthreads();
  for (int e = tid; e < 512; e += 256) {
    int h8 = e >> 6;
    float o = 0.f;
    #pragma unroll
    for (int t = 0; t < 8; ++t)
      #pragma unroll
      for (int s = 0; s < 8; ++s)
        o += attn[(h8 * 8 + t) * 8 + s] * Vm[e * 8 + s];
    osum[e] = o;
  }
  __syncthreads();
  if (tid < CH) {
    float a = 0.f;
    for (int hid = 0; hid < 512; ++hid) a += aoW[tid * 512 + hid] * osum[hid];
    float stm = 0.f;
    #pragma unroll
    for (int t = 0; t < 8; ++t) stm += ks[tid * 8 + t];
    cond[((size_t)(b * CH + tid)) * NNODE + n] = aob[tid] + 0.125f * (a + stm);
  }
}

// decoder MLP per (b,n): 32 -> 512 -> 512 -> 144
__global__ __launch_bounds__(256) void k_decoder(
    const float* __restrict__ cond,
    const float* __restrict__ d1W, const float* __restrict__ d1b,
    const float* __restrict__ d2W, const float* __restrict__ d2b,
    const float* __restrict__ d3W, const float* __restrict__ d3b,
    float* __restrict__ out) {
  __shared__ float cs[CH];
  __shared__ float y1[512];
  __shared__ float y2[512];
  int b = blockIdx.x >> 10, n = blockIdx.x & 1023;
  int tid = threadIdx.x;
  if (tid < CH) cs[tid] = cond[((size_t)(b * CH + tid)) * NNODE + n];
  __syncthreads();
  for (int e = tid; e < 512; e += 256) {
    float a = d1b[e];
    #pragma unroll
    for (int c = 0; c < CH; ++c) a += d1W[e * CH + c] * cs[c];
    y1[e] = fmaxf(a, 0.f);
  }
  __syncthreads();
  for (int e = tid; e < 512; e += 256) {
    float a = d2b[e];
    for (int k = 0; k < 512; ++k) a += d2W[e * 512 + k] * y1[k];
    y2[e] = fmaxf(a, 0.f);
  }
  __syncthreads();
  if (tid < 144) {
    float a = d3b[tid];
    for (int k = 0; k < 512; ++k) a += d3W[tid * 512 + k] * y2[k];
    out[((size_t)(b * 144 + tid)) * NNODE + n] = a;
  }
}

// =====================================================================
// host driver
// =====================================================================
static inline int cdiv(long long a, int b) { return (int)((a + b - 1) / b); }

extern "C" void kernel_launch(void* const* d_in, const int* in_sizes, int n_in,
                              void* d_out, int out_size, void* d_ws, size_t ws_size,
                              hipStream_t stream) {
  (void)in_sizes; (void)n_in; (void)out_size; (void)ws_size;

  const float* X_in     = (const float*)d_in[0];
  const float* tod_emb  = (const float*)d_in[1];
  const float* dow_emb  = (const float*)d_in[2];
  const float* spat_emb = (const float*)d_in[3];
  const float* start_W  = (const float*)d_in[4];
  const float* start_b  = (const float*)d_in[5];
  const float* res_sW   = (const float*)d_in[6];
  const float* res_sb   = (const float*)d_in[7];
  const float* res_hW   = (const float*)d_in[8];
  const float* res_hb   = (const float*)d_in[9];
  const float* skm_sW   = (const float*)d_in[10];
  const float* skm_sb   = (const float*)d_in[11];
  const float* skm_hW   = (const float*)d_in[12];
  const float* skm_hb   = (const float*)d_in[13];
  const float* nv1      = (const float*)d_in[14];
  const float* nv2      = (const float*)d_in[15];
  const float* A_static = (const float*)d_in[16];
  const float* filt_W   = (const float*)d_in[17];
  const float* filt_b   = (const float*)d_in[18];
  const float* gate_W   = (const float*)d_in[19];
  const float* gate_b   = (const float*)d_in[20];
  const float* skip_W   = (const float*)d_in[21];
  const float* skip_b   = (const float*)d_in[22];
  const float* gn_g     = (const float*)d_in[23];
  const float* gn_b     = (const float*)d_in[24];
  const float* gcn_alpha= (const float*)d_in[25];
  const float* gcn_beta = (const float*)d_in[26];
  const float* gcn_W    = (const float*)d_in[27];
  const float* gcn_b    = (const float*)d_in[28];
  const float* st_W     = (const float*)d_in[29];
  const float* st_b     = (const float*)d_in[30];
  const float* q_W      = (const float*)d_in[31];
  const float* q_b      = (const float*)d_in[32];
  const float* k_W      = (const float*)d_in[33];
  const float* k_b      = (const float*)d_in[34];
  const float* v_W      = (const float*)d_in[35];
  const float* v_b      = (const float*)d_in[36];
  const float* ao_W     = (const float*)d_in[37];
  const float* ao_b     = (const float*)d_in[38];
  const float* dec1_W   = (const float*)d_in[39];
  const float* dec1_b   = (const float*)d_in[40];
  const float* dec2_W   = (const float*)d_in[41];
  const float* dec2_b   = (const float*)d_in[42];
  const float* dec3_W   = (const float*)d_in[43];
  const float* dec3_b   = (const float*)d_in[44];
  float* out = (float*)d_out;

  // ---- workspace carve (~210 MB) ----
  char* w = (char*)d_ws;
  size_t off = 0;
  auto alloc = [&](size_t bytes) -> void* {
    void* p = w + off;
    off = (off + bytes + 255) & ~(size_t)255;
    return p;
  };
  const size_t hElems = (size_t)BATCH * CH * TMAX * NNODE;          // 8.39M
  float* h_a   = (float*)alloc(hElems * 4);
  float* h_b   = (float*)alloc(hElems * 4);
  float* h_g   = (float*)alloc(hElems * 4);
  unsigned short* hgbf = (unsigned short*)alloc(hElems * 2);
  unsigned short* xp_a = (unsigned short*)alloc((size_t)128 * TMAX * NNODE * 2);
  unsigned short* xp_b = (unsigned short*)alloc((size_t)128 * TMAX * NNODE * 2);
  float* skip_acc = (float*)alloc((size_t)BATCH * SCCH * 8 * NNODE * 4);
  float* Ebuf  = (float*)alloc((size_t)BATCH * NNODE * 32 * 4);
  float* Zadp  = (float*)alloc((size_t)NNODE * NNODE * 4);
  unsigned short* Abf_adp = (unsigned short*)alloc((size_t)NNODE * NNODE * 2);
  unsigned short* Abf_sta = (unsigned short*)alloc((size_t)NNODE * NNODE * 2);
  float* res_scale = (float*)alloc((size_t)BATCH * CH * NNODE * 4);
  float* res_shift = (float*)alloc((size_t)BATCH * CH * NNODE * 4);
  float* sk_scale  = (float*)alloc((size_t)BATCH * SCCH * NNODE * 4);
  float* sk_shift  = (float*)alloc((size_t)BATCH * SCCH * NNODE * 4);
  float* stbuf = (float*)alloc((size_t)BATCH * CH * 8 * NNODE * 4);
  float* condb = (float*)alloc((size_t)BATCH * CH * NNODE * 4);
  float* gstat = (float*)alloc(64 * 2 * 4);
  float* wbuf  = (float*)alloc(256);

  // ---- prologue ----
  {
    size_t skN = (size_t)BATCH * SCCH * 8 * NNODE;
    k_zero<<<cdiv(skN, 256), 256, 0, stream>>>(skip_acc, skN);
  }
  k_embed<<<cdiv(BATCH * NNODE, 256), 256, 0, stream>>>(X_in, tod_emb, dow_emb, spat_emb, Ebuf);
  k_f32_to_bf16<<<cdiv((long long)NNODE * NNODE, 256), 256, 0, stream>>>(A_static, Abf_sta, (size_t)NNODE * NNODE);
  k_adp_logits<<<cdiv((long long)NNODE * NNODE, 256), 256, 0, stream>>>(nv1, nv2, Zadp);
  k_row_softmax_bf16<<<NNODE, 256, 0, stream>>>(Zadp, Abf_adp);
  k_film<<<cdiv(BATCH * CH * NNODE, 256), 256, 0, stream>>>(res_sW, res_sb, Ebuf, res_scale, CH);
  k_film<<<cdiv(BATCH * CH * NNODE, 256), 256, 0, stream>>>(res_hW, res_hb, Ebuf, res_shift, CH);
  k_film<<<cdiv(BATCH * SCCH * NNODE, 256), 256, 0, stream>>>(skm_sW, skm_sb, Ebuf, sk_scale, SCCH);
  k_film<<<cdiv(BATCH * SCCH * NNODE, 256), 256, 0, stream>>>(skm_hW, skm_hb, Ebuf, sk_shift, SCCH);
  k_start<<<cdiv(BATCH * TMAX * NNODE, 256), 256, 0, stream>>>(X_in, start_W, start_b, res_scale, res_shift, h_a);

  // ---- 24 WaveNet/GCN layers ----
  float* hcur = h_a;
  float* hnext = h_b;
  int T = TMAX;
  for (int i = 0; i < NLAYER; ++i) {
    int d = 1 << (i % 3);
    int T_out = T - d;
    int nElem = BATCH * T_out * NNODE;

    k_gatedconv<<<cdiv(nElem, 256), 256, 0, stream>>>(
        hcur, filt_W + (size_t)i * CH * CH * 2, filt_b + (size_t)i * CH,
        gate_W + (size_t)i * CH * CH * 2, gate_b + (size_t)i * CH,
        h_g, hgbf, T, T_out, d);

    k_skipacc<<<cdiv(BATCH * 8 * NNODE, 256), 256, 0, stream>>>(
        h_g, skip_W + (size_t)i * SCCH * CH, skip_b + (size_t)i * SCCH, skip_acc, T_out);

    k_alphas<<<1, 1, 0, stream>>>(gcn_alpha + (size_t)i * 4, wbuf);

    dim3 ggrid(T_out, NNODE / GTN, 1);   // M/128 = T_out ; N/64 = 16
    // support 0 (static adjacency): two diffusion hops
    k_gemm_bf16<<<ggrid, 256, 0, stream>>>(hgbf, Abf_sta, xp_a);
    k_gemm_bf16<<<ggrid, 256, 0, stream>>>(xp_a, Abf_sta, xp_b);
    k_mix1<<<cdiv(nElem, 256), 256, 0, stream>>>(
        h_g, xp_a, xp_b, gcn_W + (size_t)i * CH * 160, gcn_b + (size_t)i * CH,
        wbuf, gcn_beta + i, hcur, hnext, T, T_out, d);
    // support 1 (adaptive adjacency): two diffusion hops
    k_gemm_bf16<<<ggrid, 256, 0, stream>>>(hgbf, Abf_adp, xp_a);
    k_gemm_bf16<<<ggrid, 256, 0, stream>>>(xp_a, Abf_adp, xp_b);
    k_mix2<<<cdiv(nElem, 256), 256, 0, stream>>>(
        xp_a, xp_b, gcn_W + (size_t)i * CH * 160, wbuf, hnext, T_out);

    k_gnstats<<<BATCH * 8, 256, 0, stream>>>(hnext, gstat, T_out);
    k_gnapply<<<cdiv((long long)BATCH * CH * T_out * NNODE, 256), 256, 0, stream>>>(
        hnext, gstat, gn_g + (size_t)i * CH, gn_b + (size_t)i * CH, T_out);

    float* tmp = hcur; hcur = hnext; hnext = tmp;
    T = T_out;
  }
  // T == 8 here; hcur holds final h [B,32,8,N]

  // ---- head ----
  k_stconv<<<cdiv(BATCH * 8 * NNODE, 256), 256, 0, stream>>>(
      skip_acc, sk_scale, sk_shift, st_W, st_b, stbuf);
  k_attn<<<BATCH * NNODE, 256, 0, stream>>>(
      hcur, stbuf, q_W, q_b, k_W, k_b, v_W, v_b, ao_W, ao_b, condb);
  k_decoder<<<BATCH * NNODE, 256, 0, stream>>>(
      condb, dec1_W, dec1_b, dec2_W, dec2_b, dec3_W, dec3_b, out);
}
